// StructuralSparseBlockT18a_76673756168887
// MI455X (gfx1250) — matrix operand order repaired and checked by reading, not compiled, not run on hardware
//
#include <hip/hip_runtime.h>
#include <hip/hip_bf16.h>
#include <math.h>

// ---------------------------------------------------------------------------
// Fused: hidden = x @ Wt[t]  (split-f16 WMMA, f32 accum, hidden kept in LDS)
//        -> exact radix-select top-64 (sorted desc)
//        -> y = g @ dense_W[t] + b -> exact GELU -> LayerNorm * gamma + beta
// One workgroup = 32 batch rows x 1 token. grid = (B/32, T) = (32, 32).
// Pipelining: Wt tile register-double-buffered; x tile via async LDS copy
// (GLOBAL_LOAD_ASYNC_TO_LDS_B128 + s_wait_asynccnt) when the builtin exists.
// ---------------------------------------------------------------------------

typedef _Float16 v16h __attribute__((ext_vector_type(16)));
typedef _Float16 v8h  __attribute__((ext_vector_type(8)));
typedef float    v8f  __attribute__((ext_vector_type(8)));
typedef int      v4i  __attribute__((ext_vector_type(4)));

#define AS1 __attribute__((address_space(1)))
#define AS3 __attribute__((address_space(3)))

#if defined(__has_builtin)
#if __has_builtin(__builtin_amdgcn_global_load_async_to_lds_b128) && \
    __has_builtin(__builtin_amdgcn_s_wait_asynccnt)
#define USE_ASYNC_A 1
#endif
#endif

namespace cfg {
constexpr int B  = 1024;
constexpr int L  = 1024;   // reduction dim of GEMM1
constexpr int T  = 32;     // tokens
constexpr int H  = 2048;   // hidden units
constexpr int TK = 64;     // top-k
constexpr int O  = 256;    // output units

constexpr int BM  = 32;    // rows per workgroup
constexpr int BN  = 256;   // H-chunk per workgroup iteration
constexpr int BK  = 32;    // K-step (matches wmma 16x16x32)
constexpr int LDP = 40;    // padded LDS row pitch in halfs (80B: 16B aligned, bank-spread)

// LDS layout (bytes, all 16B aligned). Staging region is dead after GEMM1,
// so top-k structures overlay it. Total = 312320 <= 320KB WGP LDS.
constexpr int OFF_AH   = 0;                        // 32*40*2      = 2560
constexpr int OFF_AL   = OFF_AH + BM * LDP * 2;    // 2560
constexpr int OFF_BH   = OFF_AL + BM * LDP * 2;    // 256*40*2     = 20480
constexpr int OFF_BL   = OFF_BH + BN * LDP * 2;    // 20480
constexpr int STAGE_END = OFF_BL + BN * LDP * 2;   // 46080
// ---- overlay (valid only after GEMM1 completes) ----
constexpr int OFF_TOPG = 0;                        // 32*64*4      = 8192
constexpr int OFF_BINS = OFF_TOPG + BM * TK * 4;   // 8*256*4      = 8192
constexpr int OFF_WSC  = OFF_BINS + 8 * 256 * 4;   // 8*8*4        = 256
static_assert(OFF_WSC + 8 * 8 * 4 <= STAGE_END, "overlay fits in staging");
// ---- persistent ----
constexpr int OFF_RAWA = STAGE_END;                // 32*32*4      = 4096 (async x tile)
constexpr int OFF_HID  = OFF_RAWA + BM * BK * 4;   // 32*2048*4    = 262144
constexpr int LDS_BYTES = OFF_HID + BM * H * 4;    // 312320
} // namespace cfg

// Order-preserving float<->uint key (descending floats == descending keys).
__device__ __forceinline__ unsigned fkey(float f) {
  unsigned u = __float_as_uint(f);
  return (u & 0x80000000u) ? ~u : (u | 0x80000000u);
}
__device__ __forceinline__ float funkey(unsigned k) {
  unsigned u = (k & 0x80000000u) ? (k & 0x7fffffffu) : ~k;
  return __uint_as_float(u);
}

__device__ __forceinline__ float wredsum(float v) {
#pragma unroll
  for (int m = 16; m > 0; m >>= 1) v += __shfl_xor(v, m, 32);
  return v;
}

// A fragment: 16x32 f16, row-major tile in LDS (pitch LDP halfs).
// ISA layout: lane L -> row L&15; lanes 0-15 hold K {0..7,16..23}, lanes 16-31 K {8..15,24..31}.
__device__ __forceinline__ v16h frag_a(const _Float16* tile, int lane) {
  const int row = lane & 15, hs = lane >> 4;
  const _Float16* p = tile + row * cfg::LDP + hs * 8;
  v8h a0 = *(const v8h*)(p);
  v8h a1 = *(const v8h*)(p + 16);
  v16h r;
#pragma unroll
  for (int i = 0; i < 8; ++i) { r[i] = a0[i]; r[i + 8] = a1[i]; }
  return r;
}

// B fragment: 32x16 (KxN) f16 from an [N][K]-transposed LDS tile (pitch LDP).
// ISA layout: lane L -> col L&15; lanes 0-15 hold K {0..7,16..23}, lanes 16-31 K {8..15,24..31}.
__device__ __forceinline__ v16h frag_b(const _Float16* tile, int lane) {
  const int n = lane & 15, hs = lane >> 4;
  const _Float16* p = tile + n * cfg::LDP + hs * 8;
  v8h a0 = *(const v8h*)(p);
  v8h a1 = *(const v8h*)(p + 16);
  v16h r;
#pragma unroll
  for (int i = 0; i < 8; ++i) { r[i] = a0[i]; r[i + 8] = a1[i]; }
  return r;
}

__device__ __forceinline__ v8f wmma16(v16h a, v16h b, v8f c) {
  // (neg_a, A, neg_b, B, c_mod, C, reuse_a, reuse_b) -> v_wmma_f32_16x16x32_f16
  return __builtin_amdgcn_wmma_f32_16x16x32_f16(false, a, false, b, (short)0, c,
                                                false, false);
}

extern "C" __global__ __launch_bounds__(256, 1)
void sparse_block_fused(const float* __restrict__ x,    // [B, L]
                        const float* __restrict__ Wt,   // [T, L, H]
                        const float* __restrict__ Wd,   // [T, TK, O]
                        const float* __restrict__ db,   // [T, O]
                        const float* __restrict__ gm,   // [T, O]
                        const float* __restrict__ bt,   // [T, O]
                        float* __restrict__ out) {      // [B, T, O]
  using namespace cfg;
  extern __shared__ char smem[];
  _Float16* Ah  = (_Float16*)(smem + OFF_AH);
  _Float16* Al  = (_Float16*)(smem + OFF_AL);
  _Float16* Bhh = (_Float16*)(smem + OFF_BH);
  _Float16* Bll = (_Float16*)(smem + OFF_BL);
  float*    hid = (float*)(smem + OFF_HID);
  float*    topg = (float*)(smem + OFF_TOPG);
#ifdef USE_ASYNC_A
  float*    rawA = (float*)(smem + OFF_RAWA);
#endif

  const int tid  = threadIdx.x;
  const int lane = tid & 31;
  const int wave = tid >> 5;
  const int bblk = blockIdx.x;       // 0..31
  const int t    = blockIdx.y;       // 0..31
  const int bBase = bblk * BM;
  const int waveN0 = wave * 32;      // wave's 32 columns inside the BN chunk

  // ======================= GEMM1: hidden = x @ Wt[t] ======================
  // Per-thread prefetch registers for the *next* K-step (hides global latency
  // under the WMMA work of the current step; LDS has no room to double-buffer).
  float4 breg[8];
#ifndef USE_ASYNC_A
  float4 areg;
#endif

  auto load_b_regs = [&](int cn, int kn) {
#pragma unroll
    for (int pass = 0; pass < 8; ++pass) {
      const int idx = pass * 1024 + tid * 4;
      const int kk = idx >> 8, n = idx & 255;
      breg[pass] =
          *(const float4*)(Wt + ((size_t)t * L + kn + kk) * H + cn * BN + n);
    }
  };
  auto issue_a = [&](int kn) {
#ifdef USE_ASYNC_A
    // Each thread async-copies one 16B chunk of the 32x32 x-tile to LDS.
    // Builtin signature (from compiler diagnostic): (AS1 v4i*, AS3 v4i*, i32, i32)
    __builtin_amdgcn_global_load_async_to_lds_b128(
        (AS1 v4i*)(x + (size_t)(bBase + (tid >> 3)) * L + kn + (tid & 7) * 4),
        (AS3 v4i*)(smem + OFF_RAWA + tid * 16), 0, 0);
#else
    areg = *(const float4*)(x + (size_t)(bBase + (tid >> 3)) * L + kn +
                            (tid & 7) * 4);
#endif
  };

  load_b_regs(0, 0);
  issue_a(0);

  for (int chunk = 0; chunk < H / BN; ++chunk) {
    v8f c00 = {}, c01 = {}, c10 = {}, c11 = {};

    for (int k0 = 0; k0 < L; k0 += BK) {
#ifdef USE_ASYNC_A
      __builtin_amdgcn_s_wait_asynccnt(0);  // x tile resident in rawA
#endif
      __syncthreads();  // staging free (frag reads done) + async A visible

      // ---- stage A: f32 -> f16 hi/lo split ----
      {
#ifdef USE_ASYNC_A
        const float4 xv = *(const float4*)(rawA + tid * 4);
#else
        const float4 xv = areg;
#endif
        const int r = tid >> 3, cc = (tid & 7) * 4;
        const float* xp = (const float*)&xv;
#pragma unroll
        for (int j = 0; j < 4; ++j) {
          const float v = xp[j];
          const _Float16 h = (_Float16)v;
          Ah[r * LDP + cc + j] = h;
          Al[r * LDP + cc + j] = (_Float16)(v - (float)h);
        }
      }
      // ---- stage B from prefetch regs (transposed to [N][K], hi/lo) ----
#pragma unroll
      for (int pass = 0; pass < 8; ++pass) {
        const int idx = pass * 1024 + tid * 4;
        const int kk = idx >> 8, n = idx & 255;
        const float* wp = (const float*)&breg[pass];
#pragma unroll
        for (int j = 0; j < 4; ++j) {
          const float v = wp[j];
          const _Float16 h = (_Float16)v;
          Bhh[(n + j) * LDP + kk] = h;
          Bll[(n + j) * LDP + kk] = (_Float16)(v - (float)h);
        }
      }
      __syncthreads();

      // ---- prefetch next step's tiles (overlaps with WMMA below) ----
      {
        int kn = k0 + BK, cn = chunk;
        if (kn == L) { kn = 0; ++cn; }
        if (cn < H / BN) {
          issue_a(kn);
          load_b_regs(cn, kn);
        }
      }

      const v16h ah0 = frag_a(Ah, lane);
      const v16h ah1 = frag_a(Ah + 16 * LDP, lane);
      const v16h al0 = frag_a(Al, lane);
      const v16h al1 = frag_a(Al + 16 * LDP, lane);
      const v16h bh0 = frag_b(Bhh + (waveN0 + 0) * LDP, lane);
      const v16h bh1 = frag_b(Bhh + (waveN0 + 16) * LDP, lane);
      const v16h bl0 = frag_b(Bll + (waveN0 + 0) * LDP, lane);
      const v16h bl1 = frag_b(Bll + (waveN0 + 16) * LDP, lane);

      // Dekker split: A*B ~= Ah*Bh + Ah*Bl + Al*Bh (f32 accumulate)
      c00 = wmma16(ah0, bh0, c00); c00 = wmma16(ah0, bl0, c00); c00 = wmma16(al0, bh0, c00);
      c01 = wmma16(ah0, bh1, c01); c01 = wmma16(ah0, bl1, c01); c01 = wmma16(al0, bh1, c01);
      c10 = wmma16(ah1, bh0, c10); c10 = wmma16(ah1, bl0, c10); c10 = wmma16(al1, bh0, c10);
      c11 = wmma16(ah1, bh1, c11); c11 = wmma16(ah1, bl1, c11); c11 = wmma16(al1, bh1, c11);
    }

    // ---- spill chunk results into the resident LDS hidden tile ----
    // C/D layout: lane L -> col L&15; VGPR i -> row i + 8*(L>>4).
    {
      const int col = chunk * BN + waveN0 + (lane & 15);
      const int rb = 8 * (lane >> 4);
#pragma unroll
      for (int i = 0; i < 8; ++i) {
        hid[(rb + i) * H + col]            = c00[i];
        hid[(rb + i) * H + col + 16]       = c01[i];
        hid[(16 + rb + i) * H + col]       = c10[i];
        hid[(16 + rb + i) * H + col + 16]  = c11[i];
      }
    }
  }
  __syncthreads();  // hidden tile complete; staging region becomes overlay

  // ============== exact top-64 per row: 4-pass radix select ===============
  unsigned* bins = (unsigned*)(smem + OFF_BINS) + wave * 256;
  unsigned* wsc  = (unsigned*)(smem + OFF_WSC) + wave * 8;

  for (int rr = 0; rr < 4; ++rr) {
    const int r = wave * 4 + rr;
    const float* hv = hid + r * H;

    unsigned prefix = 0, maskh = 0;
    int kneed = TK;
    for (int sh = 24; sh >= 0; sh -= 8) {
      for (int i = lane; i < 256; i += 32) bins[i] = 0;
      __syncthreads();
      for (int j = 0; j < H / 32; ++j) {
        const unsigned u = fkey(hv[j * 32 + lane]);
        if ((u & maskh) == prefix) atomicAdd(&bins[(u >> sh) & 255], 1u);
      }
      __syncthreads();
      if (lane == 0) {
        int cum = 0, b = 255;
        for (; b > 0; --b) {
          const int nb = cum + (int)bins[b];
          if (nb >= kneed) break;
          cum = nb;
        }
        wsc[0] = (unsigned)b;
        wsc[1] = (unsigned)cum;  // count strictly above selected bucket
      }
      __syncthreads();
      prefix |= wsc[0] << sh;
      maskh  |= 255u << sh;
      kneed  -= (int)wsc[1];
    }
    const unsigned Tkey = prefix;  // exact key of the 64th largest

    if (lane == 0) wsc[2] = 0;
    __syncthreads();
    float* tg = topg + r * TK;
    for (int j = 0; j < H / 32; ++j) {
      const float v = hv[j * 32 + lane];
      if (fkey(v) > Tkey) {
        const unsigned p = atomicAdd(&wsc[2], 1u);
        tg[p] = v;
      }
    }
    __syncthreads();
    {
      const unsigned cnt = wsc[2];  // < 64; fill remainder with the tie value
      const float tval = funkey(Tkey);
      for (unsigned p = cnt + lane; p < (unsigned)TK; p += 32) tg[p] = tval;
    }
    __syncthreads();

    // bitonic sort 64 values descending (2 per lane)
    for (int k = 2; k <= TK; k <<= 1) {
      for (int j = k >> 1; j > 0; j >>= 1) {
#pragma unroll
        for (int e = 0; e < 2; ++e) {
          const int i = lane + e * 32;
          const int p = i ^ j;
          if (p > i) {
            const float a = tg[i], bv = tg[p];
            const bool desc = ((i & k) == 0);
            if (desc ? (a < bv) : (a > bv)) { tg[i] = bv; tg[p] = a; }
          }
        }
        __syncthreads();
      }
    }
  }

  // ========== dense block (g @ W + b) -> exact GELU -> LayerNorm ==========
  const float* WD = Wd + (size_t)t * TK * O;
  float acc[4][8];
#pragma unroll
  for (int r = 0; r < 4; ++r)
#pragma unroll
    for (int i = 0; i < 8; ++i) acc[r][i] = 0.f;

  for (int k = 0; k < TK; ++k) {
    float w[8];
#pragma unroll
    for (int i = 0; i < 8; ++i) w[i] = WD[k * O + lane + 32 * i];
    float g[4];
#pragma unroll
    for (int r = 0; r < 4; ++r) g[r] = topg[(wave * 4 + r) * TK + k];
#pragma unroll
    for (int r = 0; r < 4; ++r)
#pragma unroll
      for (int i = 0; i < 8; ++i) acc[r][i] = fmaf(g[r], w[i], acc[r][i]);
  }

#pragma unroll
  for (int r = 0; r < 4; ++r) {
    float y[8];
    float s = 0.f;
#pragma unroll
    for (int i = 0; i < 8; ++i) {
      float v = acc[r][i] + db[t * O + lane + 32 * i];
      v = 0.5f * v * (1.f + erff(v * 0.70710678118654752440f));  // exact GELU
      y[i] = v;
      s += v;
    }
    s = wredsum(s);
    const float mean = s * (1.f / O);
    float s2 = 0.f;
#pragma unroll
    for (int i = 0; i < 8; ++i) {
      const float d = y[i] - mean;
      s2 += d * d;
    }
    s2 = wredsum(s2);
    const float inv = rsqrtf(s2 * (1.f / O) + 1e-6f);
    const int brow = bBase + wave * 4 + r;
#pragma unroll
    for (int i = 0; i < 8; ++i) {
      const int o = lane + 32 * i;
      out[((size_t)brow * T + t) * O + o] =
          (y[i] - mean) * inv * gm[t * O + o] + bt[t * O + o];
    }
  }
}

extern "C" void kernel_launch(void* const* d_in, const int* in_sizes, int n_in,
                              void* d_out, int out_size, void* d_ws, size_t ws_size,
                              hipStream_t stream) {
  (void)in_sizes; (void)n_in; (void)out_size; (void)d_ws; (void)ws_size;
  const float* x  = (const float*)d_in[0];
  const float* Wt = (const float*)d_in[1];
  const float* Wd = (const float*)d_in[2];
  const float* db = (const float*)d_in[3];
  const float* gm = (const float*)d_in[4];
  const float* bt = (const float*)d_in[5];
  float* out = (float*)d_out;

  dim3 grid(cfg::B / cfg::BM, cfg::T);  // (32, 32)
  dim3 block(256);                      // 8 wave32s
  sparse_block_fused<<<grid, block, cfg::LDS_BYTES, stream>>>(x, Wt, Wd, db, gm,
                                                              bt, out);
}
